// AnomalyAttention_16776142258581
// MI455X (gfx1250) — compile-verified, matrix operand order checked by reading, not run on hardware
//
#include <hip/hip_runtime.h>

typedef __attribute__((ext_vector_type(16))) __bf16 v16bf;
typedef __attribute__((ext_vector_type(8)))  float  v8f;

#define BB 2
#define LL 2048
#define HH 8
#define EE 64
#define DD 64
#define LPAD 2052   // 16-row score strip stride (floats); row byte stride 8208 = 16B-aligned
#define TM 16       // query rows per workgroup

// K-index striping for 16-bit WMMA A/B operands (CDNA5 ISA 7.12.2):
//   lanes 0-15 = group g=0, lanes 16-31 = g=1; element e of v16bf maps to
//   K = e + 8*g + (e >= 8 ? 8 : 0)
__device__ __forceinline__ int kmap(int e, int g) {
    return e + 8 * g + ((e >= 8) ? 8 : 0);
}

__launch_bounds__(256, 1)
__global__ void anomaly_attn_kernel(const float* __restrict__ q,
                                    const float* __restrict__ k,
                                    const float* __restrict__ v,
                                    const float* __restrict__ sgin,
                                    float* __restrict__ outV,
                                    float* __restrict__ outSeries,
                                    float* __restrict__ outPrior,
                                    float* __restrict__ outSigma)
{
    __shared__ float smem[TM * LPAD];   // 16 x 2052 f32 = 128.25 KB (CDNA5: 320 KB LDS/WGP)

    const int t  = blockIdx.x;          // row-tile index, 0..127
    const int h  = blockIdx.y;
    const int b  = blockIdx.z;
    const int m0 = t * TM;

    const int tid  = threadIdx.x;
    const int wave = tid >> 5;
    const int lane = tid & 31;
    const int g    = lane >> 4;         // WMMA lane group
    const int ln   = lane & 15;         // M (A) / N (B,D) index within tile

    // ---------------- Phase 1: scores strip = (1/8) * Q K^T into LDS ----------------
    v16bf qa0, qa1;                     // A-tile: Q rows m0..m0+15, K 0-31 / 32-63
    {
        const float* qrow = q + (((size_t)b * LL + (m0 + ln)) * HH + h) * EE;
        #pragma unroll
        for (int e = 0; e < 8; ++e) {
            qa0[e]     = (__bf16)qrow[      8 * g + e];
            qa0[e + 8] = (__bf16)qrow[16 +  8 * g + e];
            qa1[e]     = (__bf16)qrow[32 +  8 * g + e];
            qa1[e + 8] = (__bf16)qrow[48 +  8 * g + e];
        }
    }

    // Causality: only column tiles ct <= t contribute; strip width W = 16*(t+1).
    for (int ct = wave; ct <= t; ct += 8) {
        const int n0 = ct * 16;
        v16bf kb0, kb1;                 // B[k][n] = keys[n0+n][k]
        const float* krow = k + (((size_t)b * LL + (n0 + ln)) * HH + h) * EE;
        #pragma unroll
        for (int e = 0; e < 8; ++e) {
            kb0[e]     = (__bf16)krow[      8 * g + e];
            kb0[e + 8] = (__bf16)krow[16 +  8 * g + e];
            kb1[e]     = (__bf16)krow[32 +  8 * g + e];
            kb1[e + 8] = (__bf16)krow[48 +  8 * g + e];
        }
        v8f acc = {};
        acc = __builtin_amdgcn_wmma_f32_16x16x32_bf16(false, qa0, false, kb0,
                                                      (short)0, acc, false, false);
        acc = __builtin_amdgcn_wmma_f32_16x16x32_bf16(false, qa1, false, kb1,
                                                      (short)0, acc, false, false);
        #pragma unroll
        for (int r = 0; r < 8; ++r) {   // D: VGPR r -> row r+8g, lane -> col n
            smem[(r + 8 * g) * LPAD + n0 + ln] = acc[r] * 0.125f;  // 1/sqrt(64)
        }
    }
    __syncthreads();

    // ------------- Phase 2: causal softmax per row; emit series (b128 stores) -------
    {
        const int r = tid >> 4;         // row in tile (16 threads/row, same wave half)
        const int c = tid & 15;
        const int i = m0 + r;           // global query row; valid keys j <= i
        float* srow = smem + r * LPAD;

        // max over valid region (float4 chunks, boundary predicated; garbage beyond
        // the computed strip is excluded before fmaxf so NaNs can't leak in)
        float rmax = -3.0e38f;
        for (int j4 = c * 4; j4 <= i; j4 += 64) {
            const float4 x = *(const float4*)(srow + j4);
            rmax = fmaxf(rmax, x.x);                          // j4   <= i guaranteed
            if (j4 + 1 <= i) rmax = fmaxf(rmax, x.y);
            if (j4 + 2 <= i) rmax = fmaxf(rmax, x.z);
            if (j4 + 3 <= i) rmax = fmaxf(rmax, x.w);
        }
        #pragma unroll
        for (int m = 8; m >= 1; m >>= 1) rmax = fmaxf(rmax, __shfl_xor(rmax, m, 16));

        // exp + partial sum; tail elements of the boundary chunk are zeroed in LDS
        float rsum = 0.0f;
        for (int j4 = c * 4; j4 <= i; j4 += 64) {
            float4 x = *(const float4*)(srow + j4);
            x.x = __expf(x.x - rmax);
            x.y = (j4 + 1 <= i) ? __expf(x.y - rmax) : 0.0f;
            x.z = (j4 + 2 <= i) ? __expf(x.z - rmax) : 0.0f;
            x.w = (j4 + 3 <= i) ? __expf(x.w - rmax) : 0.0f;
            rsum += x.x + x.y + x.z + x.w;
            *(float4*)(srow + j4) = x;
        }
        #pragma unroll
        for (int m = 8; m >= 1; m >>= 1) rsum += __shfl_xor(rsum, m, 16);
        const float inv = 1.0f / rsum;

        // normalize + store: every chunk is either y*inv or all-zero -> pure b128 path
        float* gser = outSeries + (((size_t)b * HH + h) * LL + i) * LL;
        for (int j4 = c * 4; j4 < LL; j4 += 64) {
            float4 x = make_float4(0.0f, 0.0f, 0.0f, 0.0f);
            if (j4 <= i) {
                const float4 y = *(const float4*)(srow + j4);  // tail already zeroed
                x.x = y.x * inv; x.y = y.y * inv;
                x.z = y.z * inv; x.w = y.w * inv;
            }
            *(float4*)(srow + j4) = x;   // LDS == exact series row (zeros past i)
            *(float4*)(gser + j4) = x;   // coalesced global_store_b128
        }
    }
    __syncthreads();

    // ---------------- Phase 3: wave split ----------------
    if (wave < 4) {
        // V[m, d0+n] = sum_j series[m,j] * values[j, d0+n]  via bf16 WMMA, A from LDS
        const int d0 = wave * 16;
        const int W  = (t + 1) * 16;    // nonzero series width (LDS zero-padded beyond)
        v8f acc = {};
        for (int j0 = 0; j0 < W; j0 += 32) {
            v16bf a, bm;
            #pragma unroll
            for (int e = 0; e < 16; ++e) {
                const int kk = kmap(e, g);
                a[e]  = (__bf16)smem[ln * LPAD + j0 + kk];
                bm[e] = (__bf16)v[(((size_t)b * LL + (j0 + kk)) * HH + h) * DD + d0 + ln];
            }
            acc = __builtin_amdgcn_wmma_f32_16x16x32_bf16(false, a, false, bm,
                                                          (short)0, acc, false, false);
        }
        #pragma unroll
        for (int r = 0; r < 8; ++r) {
            outV[(((size_t)b * LL + (m0 + r + 8 * g)) * HH + h) * DD + d0 + ln] = acc[r];
        }
    } else {
        // prior + sigma_out: elementwise, full (unmasked) rows, b128 stores
        const int r0 = (wave - 4) * 4;
        for (int rr = 0; rr < 4; ++rr) {
            const int i = m0 + r0 + rr;
            const float s   = sgin[((size_t)b * LL + i) * HH + h];
            const float sgm = 1.0f / (1.0f + __expf(-5.0f * s)) + 1e-5f;
            const float sig = __expf(sgm * 1.0986122886681098f) - 1.0f;  // 3^sgm - 1
            const float amp = 0.3989422804014327f / sig;                 // 1/(sqrt(2pi)*sig)
            const float c2  = -0.5f / (sig * sig);
            float* gpri = outPrior + (((size_t)b * HH + h) * LL + i) * LL;
            float* gsig = outSigma + (((size_t)b * HH + h) * LL + i) * LL;
            const float4 sv = make_float4(sig, sig, sig, sig);
            for (int j4 = lane * 4; j4 < LL; j4 += 128) {
                const float d0f = (float)(i - j4);
                const float d1f = (float)(i - (j4 + 1));
                const float d2f = (float)(i - (j4 + 2));
                const float d3f = (float)(i - (j4 + 3));
                float4 p;
                p.x = amp * __expf(d0f * d0f * c2);
                p.y = amp * __expf(d1f * d1f * c2);
                p.z = amp * __expf(d2f * d2f * c2);
                p.w = amp * __expf(d3f * d3f * c2);
                *(float4*)(gpri + j4) = p;    // 512B/wave-instruction
                *(float4*)(gsig + j4) = sv;
            }
        }
    }
}

extern "C" void kernel_launch(void* const* d_in, const int* in_sizes, int n_in,
                              void* d_out, int out_size, void* d_ws, size_t ws_size,
                              hipStream_t stream) {
    const float* q = (const float*)d_in[0];
    const float* k = (const float*)d_in[1];
    const float* v = (const float*)d_in[2];
    const float* s = (const float*)d_in[3];

    float* out = (float*)d_out;
    const size_t nV = (size_t)BB * LL * HH * DD;         // 2,097,152
    const size_t nS = (size_t)BB * HH * LL * (size_t)LL; // 67,108,864
    float* outV      = out;
    float* outSeries = out + nV;
    float* outPrior  = outSeries + nS;
    float* outSigma  = outPrior + nS;

    dim3 grid(LL / TM, HH, BB);   // 128 x 8 x 2 workgroups
    anomaly_attn_kernel<<<grid, 256, 0, stream>>>(q, k, v, s,
                                                  outV, outSeries, outPrior, outSigma);
}
